// VariableRecurrent_893353197872
// MI455X (gfx1250) — matrix operand order repaired and checked by reading, not compiled, not run on hardware
//
#include <hip/hip_runtime.h>
#include <hip/hip_bf16.h>

// Problem constants (match reference)
#define TT     2048
#define BB     32
#define II     256
#define HH     256
#define G3     768
#define TOTAL  49664          // sum of lengths
#define MTILES 3104           // TOTAL / 16 (exact)

typedef __attribute__((ext_vector_type(16))) __bf16 v16bf;
typedef __attribute__((ext_vector_type(8)))  __bf16 bf16x8;
typedef __attribute__((ext_vector_type(4)))  __bf16 bf16x4;
typedef __attribute__((ext_vector_type(8)))  float  v8f;
typedef __attribute__((ext_vector_type(4)))  float  f32x4;

__device__ __forceinline__ v8f wmma_bf16(v16bf a, v16bf b, v8f c) {
  // D = A(16x32 bf16) * B(32x16 bf16) + C(16x16 f32)
  return __builtin_amdgcn_wmma_f32_16x16x32_bf16(false, a, false, b, (short)0, c,
                                                 false, false);
}

__device__ __forceinline__ v16bf cat16(bf16x8 lo, bf16x8 hi) {
  return __builtin_shufflevector(lo, hi, 0, 1, 2, 3, 4, 5, 6, 7,
                                 8, 9, 10, 11, 12, 13, 14, 15);
}

__device__ __forceinline__ v16bf cvt16(const float* p) {
  v16bf r;
#pragma unroll
  for (int e = 0; e < 16; ++e) r[e] = (__bf16)p[e];
  return r;
}

__device__ __forceinline__ float sigmoid_f(float x) {
  return 1.0f / (1.0f + __expf(-x));
}
__device__ __forceinline__ float tanh_f(float x) {
  float e = __expf(-2.0f * x);
  return (1.0f - e) / (1.0f + e);
}

// ---------------------------------------------------------------- init / cvt
__global__ void init_ctr(unsigned* ctr) { *ctr = 0u; }

__global__ void cvt_f32_bf16(const float* __restrict__ src,
                             __bf16* __restrict__ dst, int n4) {
  int i = blockIdx.x * blockDim.x + threadIdx.x;
  if (i < n4) {
    f32x4 v = ((const f32x4*)src)[i];
    bf16x4 o;
    o.x = (__bf16)v.x; o.y = (__bf16)v.y; o.z = (__bf16)v.z; o.w = (__bf16)v.w;
    ((bf16x4*)dst)[i] = o;
  }
}

// --------------------------------------------------- gx = X @ W_ih^T + b_ih
// grid.x = MTILES (one 16-row tile), block = 512 (16 waves x 3 n-tiles = 48)
__global__ __launch_bounds__(512) void gx_gemm(const __bf16* __restrict__ X,
                                               const __bf16* __restrict__ W,
                                               const float* __restrict__ b_ih,
                                               float* __restrict__ gx) {
  int tid  = threadIdx.x;
  int wave = tid >> 5, lane = tid & 31;
  int l15  = lane & 15, lhi = lane >> 4;
  int m0   = blockIdx.x * 16;
  int nb   = wave * 3;

  const __bf16* ar  = X + (size_t)(m0 + l15) * II;
  const __bf16* w0r = W + (size_t)((nb + 0) * 16 + l15) * II;
  const __bf16* w1r = W + (size_t)((nb + 1) * 16 + l15) * II;
  const __bf16* w2r = W + (size_t)((nb + 2) * 16 + l15) * II;

  v8f a0 = {}, a1 = {}, a2 = {};
#pragma unroll
  for (int k = 0; k < 8; ++k) {
    int ka = k * 32 + lhi * 16;                        // A frag: 16 contiguous K
    v16bf av = cat16(*(const bf16x8*)(ar + ka), *(const bf16x8*)(ar + ka + 8));
    int kb = k * 32 + lhi * 16;                        // B frag: 16 contiguous K
    a0 = wmma_bf16(av, cat16(*(const bf16x8*)(w0r + kb), *(const bf16x8*)(w0r + kb + 8)), a0);
    a1 = wmma_bf16(av, cat16(*(const bf16x8*)(w1r + kb), *(const bf16x8*)(w1r + kb + 8)), a1);
    a2 = wmma_bf16(av, cat16(*(const bf16x8*)(w2r + kb), *(const bf16x8*)(w2r + kb + 8)), a2);
  }
  int c0 = (nb + 0) * 16 + l15, c1 = c0 + 16, c2 = c0 + 32;
  float bb0 = b_ih[c0], bb1 = b_ih[c1], bb2 = b_ih[c2];
#pragma unroll
  for (int i = 0; i < 8; ++i) {
    size_t row = (size_t)(m0 + i + lhi * 8) * G3;      // C layout: M = i + (lane>>4)*8
    gx[row + c0] = a0[i] + bb0;
    gx[row + c1] = a1[i] + bb1;
    gx[row + c2] = a2[i] + bb2;
  }
}

// ------------------------------------------------------------- GRU scan
// 2 persistent workgroups (wg = column half), 512 threads = 16 waves.
// Wave w: m-tile = w&1 (batch rows), j-block = w>>1 (16 h-columns), owns the
// (r,z,n) WMMA tile triple for those columns; W_hh B-fragments live in VGPRs.
__global__ __launch_bounds__(512) void gru_scan(const float* __restrict__ gx,
                                                const float* __restrict__ h0,
                                                const float* __restrict__ W_hh,
                                                const float* __restrict__ b_hh,
                                                const int* __restrict__ batch_sizes,
                                                float* __restrict__ hfin,
                                                float* __restrict__ outp,
                                                __bf16* __restrict__ exch,
                                                unsigned* __restrict__ ctr) {
  __shared__ __bf16 hbuf[32][264];     // full h, bf16, padded rows (A fragments)
  __shared__ float  hstate[32 * 132];  // own 128 columns, f32, padded rows

  int tid  = threadIdx.x;
  int wg   = blockIdx.x;               // 0 or 1
  int wave = tid >> 5, lane = tid & 31;
  int l15  = lane & 15, lhi = lane >> 4;
  int m    = wave & 1, jb = wave >> 1;
  int jloc = jb * 16 + l15;            // 0..127 (this lane's column)
  int colR = wg * 128 + jloc;          // global gate columns
  int colZ = 256 + colR;
  int colN = 512 + colR;

  // ---- init h from h0
  {
    int b = tid >> 4, c0 = (tid & 15) * 16;
#pragma unroll
    for (int e = 0; e < 16; ++e) {
      float v = h0[b * 256 + c0 + e];
      hbuf[b][c0 + e] = (__bf16)v;
      int jl = c0 + e - wg * 128;
      if (jl >= 0 && jl < 128) hstate[b * 132 + jl] = v;
    }
  }

  // ---- preload W_hh B-fragments into registers (192 VGPRs/wave), bf16
  v16bf Bf[3][8];
#pragma unroll
  for (int g = 0; g < 3; ++g) {
    const float* wr = W_hh + (size_t)(g * 256 + wg * 128 + jloc) * HH;
#pragma unroll
    for (int k = 0; k < 8; ++k) Bf[g][k] = cvt16(wr + k * 32 + lhi * 16);
  }
  float bhr = b_hh[colR], bhz = b_hh[colZ], bhn = b_hh[colN];
  __syncthreads();

  unsigned off = 0;
  for (int t = 0; t < TT; ++t) {
    int bs = batch_sizes[t];

    // C-operand init: gx(+b_ih) + b_hh for r/z; b_hh only for n
    v8f accR, accZ, accN;
#pragma unroll
    for (int i = 0; i < 8; ++i) {
      int b = m * 16 + lhi * 8 + i;
      unsigned row = off + (unsigned)b;
      if (row >= TOTAL) row = TOTAL - 1;
      const float* gr = gx + (size_t)row * G3;
      accR[i] = gr[colR] + bhr;
      accZ[i] = gr[colZ] + bhz;
      accN[i] = bhn;
    }
    // prefetch next step's gx rows while WMMAs run
    {
      unsigned nrow = off + (unsigned)bs + (unsigned)(m * 16 + lhi * 8);
      if (nrow >= TOTAL) nrow = TOTAL - 1;
      __builtin_prefetch(gx + (size_t)nrow * G3 + colR, 0, 3);
    }

    // gh accumulation: A from LDS (2x ds_load_b128 / frag), B resident
#pragma unroll
    for (int k = 0; k < 8; ++k) {
      const __bf16* hp = &hbuf[m * 16 + l15][k * 32 + lhi * 16];
      v16bf av = cat16(*(const bf16x8*)hp, *(const bf16x8*)(hp + 8));
      accR = wmma_bf16(av, Bf[0][k], accR);
      accZ = wmma_bf16(av, Bf[1][k], accZ);
      accN = wmma_bf16(av, Bf[2][k], accN);
    }

    // gates in registers (each lane owns its (b, j) elements)
    float hn[8];
#pragma unroll
    for (int i = 0; i < 8; ++i) {
      int b = m * 16 + lhi * 8 + i;
      float hold = hstate[b * 132 + jloc];
      unsigned row = off + (unsigned)b;
      if (row >= TOTAL) row = TOTAL - 1;
      float gxn = gx[(size_t)row * G3 + colN];
      float r = sigmoid_f(accR[i]);
      float z = sigmoid_f(accZ[i]);
      float n = tanh_f(fmaf(r, accN[i], gxn));
      float h = (1.0f - z) * n + z * hold;
      hn[i] = (b < bs) ? h : hold;   // freeze ended sequences
    }

    __syncthreads();  // all A-fragment reads done before h is overwritten
    unsigned par = (unsigned)(t & 1);
    __bf16* exw = exch + (size_t)(par * 2 + (unsigned)wg) * 4096;
#pragma unroll
    for (int i = 0; i < 8; ++i) {
      int b = m * 16 + lhi * 8 + i;
      hstate[b * 132 + jloc] = hn[i];
      hbuf[b][colR] = (__bf16)hn[i];
      exw[b * 128 + jloc] = (__bf16)hn[i];
      if (b < bs) outp[(size_t)(off + (unsigned)b) * HH + colR] = hn[i];
    }

    // ---- cross-workgroup barrier (monotonic counter, L2 scope)
    __threadfence();
    __syncthreads();
    if (tid == 0) {
      __hip_atomic_fetch_add(ctr, 1u, __ATOMIC_RELEASE, __HIP_MEMORY_SCOPE_AGENT);
      unsigned target = 2u * (unsigned)(t + 1);
      while (__hip_atomic_load(ctr, __ATOMIC_ACQUIRE, __HIP_MEMORY_SCOPE_AGENT) < target)
        __builtin_amdgcn_s_sleep(1);
    }
    __syncthreads();
    __threadfence();   // invalidate stale lines before reading peer's half

    // pull the other workgroup's h half into hbuf
    {
      const __bf16* exr = exch + (size_t)(par * 2 + (unsigned)(1 - wg)) * 4096;
      int b = tid >> 4, cg = tid & 15;
      bf16x8 v = *(const bf16x8*)(exr + b * 128 + cg * 8);
      *(bf16x8*)(&hbuf[b][(1 - wg) * 128 + cg * 8]) = v;
    }
    __syncthreads();
    off += (unsigned)bs;
  }

  // ---- final hidden state (own half)
  {
    int b = tid >> 4, c0 = (tid & 15) * 8;
#pragma unroll
    for (int e = 0; e < 8; ++e)
      hfin[b * 256 + wg * 128 + c0 + e] = hstate[b * 132 + c0 + e];
  }
}

// ---------------------------------------------------------------- launcher
extern "C" void kernel_launch(void* const* d_in, const int* in_sizes, int n_in,
                              void* d_out, int out_size, void* d_ws, size_t ws_size,
                              hipStream_t stream) {
  (void)in_sizes; (void)n_in; (void)out_size; (void)ws_size;
  const float* input_ = (const float*)d_in[0];
  const float* h0     = (const float*)d_in[1];
  const float* W_ih   = (const float*)d_in[2];
  const float* W_hh   = (const float*)d_in[3];
  const float* b_ih   = (const float*)d_in[4];
  const float* b_hh   = (const float*)d_in[5];
  const int*   bsizes = (const int*)d_in[6];

  float* out  = (float*)d_out;
  float* hfin = out;               // [32, 256]
  float* outp = out + 32 * 256;    // [TOTAL, 256]

  // workspace layout (gx first so a clamped tail over-read stays in-bounds)
  char*  ws      = (char*)d_ws;
  size_t offGx   = 0;
  size_t offXbf  = offGx  + (size_t)TOTAL * G3 * 4;   // 152.6 MB
  size_t offWbf  = offXbf + (size_t)TOTAL * II * 2;   // +25.4 MB
  size_t offExch = offWbf + (size_t)G3 * II * 2;      // +384 KB
  size_t offCtr  = offExch + 2 * 2 * 4096 * 2;        // +32 KB

  float*    gx   = (float*)(ws + offGx);
  __bf16*   Xbf  = (__bf16*)(ws + offXbf);
  __bf16*   Wbf  = (__bf16*)(ws + offWbf);
  __bf16*   exch = (__bf16*)(ws + offExch);
  unsigned* ctr  = (unsigned*)(ws + offCtr);

  init_ctr<<<1, 1, 0, stream>>>(ctr);

  int n4x = TOTAL * II / 4;
  cvt_f32_bf16<<<(n4x + 255) / 256, 256, 0, stream>>>(input_, Xbf, n4x);
  int n4w = G3 * II / 4;
  cvt_f32_bf16<<<(n4w + 255) / 256, 256, 0, stream>>>(W_ih, Wbf, n4w);

  gx_gemm<<<MTILES, 512, 0, stream>>>(Xbf, Wbf, b_ih, gx);
  gru_scan<<<2, 512, 0, stream>>>(gx, h0, W_hh, b_hh, bsizes, hfin, outp, exch, ctr);
}
